// MultiHeadSelfAttention_20418274525478
// MI455X (gfx1250) — compile-verified, matrix operand order checked by reading, not run on hardware
//
#include <hip/hip_runtime.h>
#include <math.h>

// ---------------------------------------------------------------------------
// MHA for MI455X (gfx1250, wave32, WMMA).  B=2 L=2048 D=2048 H=16 DH=128.
// Pipeline: [gemm Q/K/VT in bf16] -> [flash attention, bf16 WMMA + f32 softmax,
//            async double-buffered K/V tiles] -> [gemm @ W_o^T -> f32 out]
// ---------------------------------------------------------------------------

namespace {

constexpr int Bn  = 2;
constexpr int Ln  = 2048;
constexpr int Dn  = 2048;
constexpr int Hn  = 16;
constexpr int DHn = 128;
constexpr int Mn  = Bn * Ln;                       // 4096 GEMM rows
constexpr float kScale = 0.08838834764831845f;     // 1/sqrt(128)

typedef __bf16 bf16_t;
typedef __attribute__((ext_vector_type(16))) __bf16        v16bf;
typedef __attribute__((ext_vector_type(8)))  float         v8f;
typedef __attribute__((ext_vector_type(4)))  unsigned int  u32x4;

union Frag16 { v16bf v; u32x4 u[2]; };

// A/B fragment for v_wmma_f32_16x16x32_bf16, loaded from a row-major tile.
// lane (0..31): row = caller-selected (lane&15), hsel = lane>>4.
// elements 0..7  = K[hsel*8 .. hsel*8+7], elements 8..15 = K[16+hsel*8 ..].
__device__ __forceinline__ v16bf frag_load(const bf16_t* row, int hsel) {
  Frag16 f;
  f.u[0] = *(const u32x4*)(row + hsel * 8);
  f.u[1] = *(const u32x4*)(row + 16 + hsel * 8);
  return f.v;
}

__device__ __forceinline__ v8f wmma_bf16(v16bf a, v16bf b, v8f c) {
  return __builtin_amdgcn_wmma_f32_16x16x32_bf16(
      /*neg_a=*/false, a, /*neg_b=*/false, b,
      /*c_mod=*/(short)0, c, /*reuse_a=*/false, /*reuse_b=*/false);
}

// Async HBM -> LDS copy, 16 bytes per lane, tracked by ASYNCcnt (gfx1250).
// LDS dest offset = low 32 bits of the generic pointer (LDS aperture rule).
__device__ __forceinline__ void async_copy_b128(void* lds_dst, const void* gsrc) {
  const unsigned off = (unsigned)(unsigned long long)lds_dst;
  asm volatile("global_load_async_to_lds_b128 %0, %1, off"
               :: "v"(off), "v"(gsrc) : "memory");
}
__device__ __forceinline__ void wait_async0() {
  asm volatile("s_wait_asynccnt 0" ::: "memory");
}

// ---------------------------------------------------------------------------
// Generic GEMM: Y[m][n] = alpha * sum_k A[m][k] * W[n][k]   (i.e. A @ W^T)
// A: f32 or bf16 row-major [Mn][Dn]; W: f32 row-major [Dn][Dn].
// MODE 0: bf16 out, head layout [B][H][L][DH]
// MODE 1: bf16 out, transposed head layout [B][H][DH][L]   (for V)
// MODE 2: f32 out, row-major [Mn][Dn]                      (final output)
// Block: 256 threads = 8 waves; macro tile 256(M)x128(N); wave tile 64x64:
// per K-step 16 WMMAs vs 8 LDS fragment loads.
// ---------------------------------------------------------------------------
template <typename AT, int MODE>
__global__ __launch_bounds__(256) void gemm_bf16_kernel(
    const AT* __restrict__ A, const float* __restrict__ W,
    void* __restrict__ out, float alpha) {
  __shared__ alignas(16) bf16_t sA[256 * 40];   // [m-row][k], pad->40 (80B rows)
  __shared__ alignas(16) bf16_t sB[128 * 40];   // [n-row of W][k]

  const int tid  = threadIdx.x;
  const int lane = tid & 31;
  const int wid  = tid >> 5;
  const int wm   = wid & 3;        // wave M index: 4 waves down  (64 rows each)
  const int wn   = wid >> 2;       // wave N index: 2 waves across (64 cols each)
  const int m0   = blockIdx.y * 256;
  const int n0   = blockIdx.x * 128;
  const int l16  = lane & 15;
  const int hs   = lane >> 4;

  v8f acc[4][4];
#pragma unroll
  for (int mt = 0; mt < 4; ++mt)
#pragma unroll
    for (int nt = 0; nt < 4; ++nt)
      acc[mt][nt] = v8f{0.f, 0.f, 0.f, 0.f, 0.f, 0.f, 0.f, 0.f};

  for (int k0 = 0; k0 < Dn; k0 += 32) {
    // --- cooperative LDS fill, converting to bf16 on the fly
#pragma unroll
    for (int it = 0; it < 8; ++it) {           // A: 256 rows x 32 k
      const int idx = tid + it * 256;          // 0..2047
      const int row = idx >> 3;                // 0..255
      const int col = (idx & 7) << 2;          // 0,4,...,28
      const AT* src = A + (size_t)(m0 + row) * Dn + (k0 + col);
      bf16_t* dst = &sA[row * 40 + col];
      if constexpr (sizeof(AT) == 4) {
        const float4 t = *(const float4*)src;
        dst[0] = (bf16_t)t.x; dst[1] = (bf16_t)t.y;
        dst[2] = (bf16_t)t.z; dst[3] = (bf16_t)t.w;
      } else {
        *(unsigned long long*)dst = *(const unsigned long long*)src;
      }
    }
#pragma unroll
    for (int it = 0; it < 4; ++it) {           // W: 128 rows x 32 k
      const int idx = tid + it * 256;          // 0..1023
      const int row = idx >> 3;                // 0..127
      const int col = (idx & 7) << 2;
      const float4 t = *(const float4*)(W + (size_t)(n0 + row) * Dn + (k0 + col));
      bf16_t* dst = &sB[row * 40 + col];
      dst[0] = (bf16_t)t.x; dst[1] = (bf16_t)t.y;
      dst[2] = (bf16_t)t.z; dst[3] = (bf16_t)t.w;
    }
    if (k0 + 32 < Dn) {   // prefetch next K-slab (global_prefetch_b8)
      __builtin_prefetch(A + (size_t)(m0 + tid) * Dn + (k0 + 32), 0, 0);
      __builtin_prefetch(W + (size_t)(n0 + (tid >> 1)) * Dn + (k0 + 32), 0, 0);
    }
    __syncthreads();

    v16bf afr[4], bfr[4];
#pragma unroll
    for (int mt = 0; mt < 4; ++mt)
      afr[mt] = frag_load(&sA[(wm * 64 + mt * 16 + l16) * 40], hs);
#pragma unroll
    for (int nt = 0; nt < 4; ++nt)
      bfr[nt] = frag_load(&sB[(wn * 64 + nt * 16 + l16) * 40], hs);
#pragma unroll
    for (int mt = 0; mt < 4; ++mt)
#pragma unroll
      for (int nt = 0; nt < 4; ++nt)
        acc[mt][nt] = wmma_bf16(afr[mt], bfr[nt], acc[mt][nt]);
    __syncthreads();
  }

  // --- store C fragments. lane: n = tile_n + l16, m = tile_m + hs*8 + r.
#pragma unroll
  for (int mt = 0; mt < 4; ++mt)
#pragma unroll
    for (int nt = 0; nt < 4; ++nt)
#pragma unroll
      for (int r = 0; r < 8; ++r) {
        const int m = m0 + wm * 64 + mt * 16 + hs * 8 + r;
        const int n = n0 + wn * 64 + nt * 16 + l16;
        const float v = alpha * acc[mt][nt][r];
        if constexpr (MODE == 2) {
          ((float*)out)[(size_t)m * Dn + n] = v;
        } else {
          const int bb = m >> 11, ll = m & (Ln - 1);
          const int hh = n >> 7,  dh = n & (DHn - 1);
          bf16_t* o = (bf16_t*)out;
          if constexpr (MODE == 0)
            o[(((size_t)bb * Hn + hh) * Ln + ll) * DHn + dh] = (bf16_t)v;
          else
            o[(((size_t)bb * Hn + hh) * DHn + dh) * Ln + ll] = (bf16_t)v;
        }
      }
}

// ---------------------------------------------------------------------------
// Flash attention (causal).  Q: [B,H,L,DH] bf16 (pre-scaled by 1/sqrt(DH)),
// K: [B,H,L,DH] bf16, VT: [B,H,DH,L] bf16.  AO out: [B,L,D] bf16.
// Block = 128 threads (4 waves); block handles 64 query rows of one (b,h);
// each wave owns 16 rows.  32-key K/VT tiles are double-buffered in LDS via
// GLOBAL_LOAD_ASYNC_TO_LDS_B128 (ASYNCcnt), overlapping DMA with WMMA.
// ---------------------------------------------------------------------------
__global__ __launch_bounds__(128) void flash_attn_kernel(
    const bf16_t* __restrict__ Q, const bf16_t* __restrict__ K,
    const bf16_t* __restrict__ VT, bf16_t* __restrict__ AO) {
  __shared__ alignas(16) bf16_t sK[2][32 * 136];    // [key][dh], pad 8
  __shared__ alignas(16) bf16_t sV[2][128 * 40];    // [dh][key], pad 8
  __shared__ alignas(16) bf16_t sP[4][16 * 40];     // per-wave P relayout

  const int bh   = blockIdx.y;
  const int b    = bh >> 4;
  const int h    = bh & 15;
  const int q0   = blockIdx.x * 64;
  const int tid  = threadIdx.x;
  const int lane = tid & 31;
  const int wid  = tid >> 5;
  const int wq0  = q0 + wid * 16;       // this wave's first query row
  const int l16  = lane & 15;
  const int hs   = lane >> 4;

  const size_t hbase = (size_t)bh * Ln * DHn;
  const bf16_t* Qh  = Q + hbase;
  const bf16_t* Kh  = K + hbase;
  const bf16_t* VTh = VT + hbase;

  // Per-thread piece of one tile's async fill (4 b128 for K, 4 for VT).
  auto issue_tile = [&](int k0, int buf) {
#pragma unroll
    for (int it = 0; it < 4; ++it) {
      const int idx = tid + it * 128;          // 0..511
      {
        const int row = idx >> 4, c = (idx & 15) * 8;
        async_copy_b128(&sK[buf][row * 136 + c],
                        Kh + (size_t)(k0 + row) * DHn + c);
      }
      {
        const int row = idx >> 2, c = (idx & 3) * 8;
        async_copy_b128(&sV[buf][row * 40 + c],
                        VTh + (size_t)row * Ln + (k0 + c));
      }
    }
  };

  // Q fragments for this wave's 16 rows: 4 chunks over DH=128.
  v16bf qf[4];
  {
    const bf16_t* qrow = Qh + (size_t)(wq0 + l16) * DHn;
#pragma unroll
    for (int kc = 0; kc < 4; ++kc) qf[kc] = frag_load(qrow + kc * 32, hs);
  }

  v8f o[8];
#pragma unroll
  for (int f = 0; f < 8; ++f) o[f] = v8f{0.f,0.f,0.f,0.f,0.f,0.f,0.f,0.f};
  float mrow[8], lrow[8];
#pragma unroll
  for (int r = 0; r < 8; ++r) { mrow[r] = -1e30f; lrow[r] = 0.f; }

  const int kend = q0 + 64;
  int buf = 0;
  issue_tile(0, 0);
  for (int k0 = 0; k0 < kend; k0 += 32, buf ^= 1) {
    wait_async0();        // my copies for tile k0 have landed
    __syncthreads();      // everyone's copies landed; prev compute finished
    if (k0 + 32 < kend) issue_tile(k0 + 32, buf ^ 1);   // overlap with compute

    if (k0 <= wq0 + 15) {   // wave-uniform: skip fully-masked tiles
      const bf16_t* tK = &sK[buf][0];
      const bf16_t* tV = &sV[buf][0];
      // --- S = Q K^T: 16q x 32keys as two 16x16 halves, reduced over DH
      v8f s0 = v8f{0.f,0.f,0.f,0.f,0.f,0.f,0.f,0.f};
      v8f s1 = v8f{0.f,0.f,0.f,0.f,0.f,0.f,0.f,0.f};
#pragma unroll
      for (int kc = 0; kc < 4; ++kc) {
        v16bf kb0 = frag_load(&tK[(l16)      * 136 + kc * 32], hs);
        v16bf kb1 = frag_load(&tK[(16 + l16) * 136 + kc * 32], hs);
        s0 = wmma_bf16(qf[kc], kb0, s0);
        s1 = wmma_bf16(qf[kc], kb1, s1);
      }

      // --- causal mask (only tiles crossing the diagonal)
      const int rbase = wq0 + hs * 8;
      if (k0 + 31 > wq0) {
        const int c0 = k0 + l16, c1 = c0 + 16;
#pragma unroll
        for (int r = 0; r < 8; ++r) {
          const int row = rbase + r;
          if (c0 > row) s0[r] = -1e30f;
          if (c1 > row) s1[r] = -1e30f;
        }
      }

      // --- online softmax: rows live across the 16 lanes of each half-wave
      float rmax[8], rsum[8], al[8];
#pragma unroll
      for (int r = 0; r < 8; ++r) rmax[r] = fmaxf(s0[r], s1[r]);
#pragma unroll
      for (int off = 1; off < 16; off <<= 1)
#pragma unroll
        for (int r = 0; r < 8; ++r)
          rmax[r] = fmaxf(rmax[r], __shfl_xor(rmax[r], off, 32));
#pragma unroll
      for (int r = 0; r < 8; ++r) {
        const float mnew = fmaxf(mrow[r], rmax[r]);
        al[r] = __expf(mrow[r] - mnew);
        mrow[r] = mnew;
        s0[r] = __expf(s0[r] - mnew);
        s1[r] = __expf(s1[r] - mnew);
        rsum[r] = s0[r] + s1[r];
      }
#pragma unroll
      for (int off = 1; off < 16; off <<= 1)
#pragma unroll
        for (int r = 0; r < 8; ++r)
          rsum[r] += __shfl_xor(rsum[r], off, 32);
#pragma unroll
      for (int r = 0; r < 8; ++r) lrow[r] = lrow[r] * al[r] + rsum[r];
#pragma unroll
      for (int f = 0; f < 8; ++f)
#pragma unroll
        for (int r = 0; r < 8; ++r) o[f][r] *= al[r];

      // --- relayout P (C-layout -> A-layout) through per-wave LDS
      bf16_t* pw = &sP[wid][0];
#pragma unroll
      for (int r = 0; r < 8; ++r) {
        const int prow = hs * 8 + r;
        pw[prow * 40 + l16]      = (bf16_t)s0[r];
        pw[prow * 40 + 16 + l16] = (bf16_t)s1[r];
      }
      // same-wave DS ops are processed in order; just drain the counter
      asm volatile("s_wait_dscnt 0" ::: "memory");
      const v16bf pf = frag_load(&pw[l16 * 40], hs);

      // --- O += P @ V  (8 dh-tiles of 16)
#pragma unroll
      for (int f = 0; f < 8; ++f) {
        v16bf vb = frag_load(&tV[(f * 16 + l16) * 40], hs);
        o[f] = wmma_bf16(pf, vb, o[f]);
      }
    }
  }

  // --- normalize and store to AO [B][L][D] bf16
#pragma unroll
  for (int r = 0; r < 8; ++r) lrow[r] = 1.0f / lrow[r];
  const int rowb = wq0 + hs * 8;
#pragma unroll
  for (int f = 0; f < 8; ++f)
#pragma unroll
    for (int r = 0; r < 8; ++r) {
      const size_t idx =
          ((size_t)b * Ln + (rowb + r)) * Dn + h * DHn + f * 16 + l16;
      AO[idx] = (bf16_t)(o[f][r] * lrow[r]);
    }
}

}  // namespace

// ---------------------------------------------------------------------------
extern "C" void kernel_launch(void* const* d_in, const int* in_sizes, int n_in,
                              void* d_out, int out_size, void* d_ws,
                              size_t ws_size, hipStream_t stream) {
  (void)in_sizes; (void)n_in; (void)out_size; (void)ws_size;
  const float* x  = (const float*)d_in[0];
  const float* Wq = (const float*)d_in[1];
  const float* Wk = (const float*)d_in[2];
  const float* Wv = (const float*)d_in[3];
  const float* Wo = (const float*)d_in[4];

  char* ws = (char*)d_ws;
  const size_t seg = (size_t)Bn * Hn * Ln * DHn * sizeof(bf16_t);  // 16 MB
  bf16_t* Qb  = (bf16_t*)(ws);
  bf16_t* Kb  = (bf16_t*)(ws + seg);
  bf16_t* VTb = (bf16_t*)(ws + 2 * seg);
  bf16_t* AO  = (bf16_t*)(ws + 3 * seg);

  const dim3 gg(Dn / 128, Mn / 256);   // (16, 16)
  const dim3 gb(256);
  gemm_bf16_kernel<float, 0><<<gg, gb, 0, stream>>>(x, Wq, Qb, kScale);
  gemm_bf16_kernel<float, 0><<<gg, gb, 0, stream>>>(x, Wk, Kb, 1.0f);
  gemm_bf16_kernel<float, 1><<<gg, gb, 0, stream>>>(x, Wv, VTb, 1.0f);

  const dim3 ga(Ln / 64, Bn * Hn);     // (32, 32)
  flash_attn_kernel<<<ga, dim3(128), 0, stream>>>(Qb, Kb, VTb, AO);

  gemm_bf16_kernel<bf16_t, 2><<<gg, gb, 0, stream>>>(AO, Wo, d_out, 1.0f);
}
